// GroupedGRULayer_new_51840255262795
// MI455X (gfx1250) — matrix operand order repaired and checked by reading, not compiled, not run on hardware
//
#include <hip/hip_runtime.h>
#include <cstdint>

// Problem constants (match reference)
#define B_    64
#define T_    500
#define IN_   1024
#define HID_  1024
#define G_    8
#define IG_   128
#define HG_   128
#define H3_   384
#define BTILE 16

typedef __attribute__((ext_vector_type(16))) __bf16 v16bf;
typedef __attribute__((ext_vector_type(8)))  __bf16 v8bf;
typedef __attribute__((ext_vector_type(8)))  float  v8f;
typedef __attribute__((ext_vector_type(4)))  float  v4f;

// round-to-nearest-even fp32 -> bf16, packed pair into one dword
static __device__ __forceinline__ uint32_t pk2(float a, float b) {
  uint32_t ua = __float_as_uint(a); ua += 0x7FFFu + ((ua >> 16) & 1u);
  uint32_t ub = __float_as_uint(b); ub += 0x7FFFu + ((ub >> 16) & 1u);
  return (ua >> 16) | (ub & 0xFFFF0000u);
}
static __device__ __forceinline__ __bf16 f2bf(float f) {
  uint32_t u = __float_as_uint(f); u += 0x7FFFu + ((u >> 16) & 1u);
  unsigned short s = (unsigned short)(u >> 16);
  return __builtin_bit_cast(__bf16, s);
}
static __device__ __forceinline__ float sigm(float x) {
  return 1.0f / (1.0f + __expf(-x));
}
static __device__ __forceinline__ float tanh_fast(float x) {
  float ax = fabsf(x);
  float e  = __expf(2.0f * ax);
  float t  = 1.0f - 2.0f / (e + 1.0f);
  return copysignf(t, x);
}

// One workgroup = (16-batch tile) x (one group g). 8 waves; wave w owns hidden
// columns [16w, 16w+16). Recurrence over t with per-step workgroup barriers.
__global__ __launch_bounds__(256, 1) void grouped_gru_fused(
    const float* __restrict__ x,     // [B,T,IN]
    const float* __restrict__ W_ih,  // [G,384,128]
    const float* __restrict__ W_hh,  // [G,384,128]
    const float* __restrict__ b_ih,  // [G,384]
    const float* __restrict__ b_hh,  // [G,384]
    float* __restrict__ out)         // [B,T,HID]
{
  __shared__ __align__(16) __bf16 lds_wih[H3_ * IG_]; // 98304 B, resident all 500 steps
  __shared__ __align__(16) __bf16 lds_h[BTILE * HG_]; // 4 KB, h state (bf16 mirror)
  __shared__ __align__(16) __bf16 lds_x[BTILE * IG_]; // 4 KB, current x tile

  const int tid  = threadIdx.x;
  const int lane = tid & 31;
  const int w    = tid >> 5;
  const int g    = blockIdx.x & (G_ - 1);
  const int b0   = (blockIdx.x >> 3) * BTILE;

  // ---- one-time init: stage W_ih[g] -> LDS as bf16 -------------------------
  {
    const float* Wi = W_ih + (size_t)g * H3_ * IG_;
    uint32_t* dst = (uint32_t*)lds_wih;
    for (int i = tid * 4; i < H3_ * IG_; i += 256 * 4) {
      v4f v = *(const v4f*)(Wi + i);
      uint2 p; p.x = pk2(v[0], v[1]); p.y = pk2(v[2], v[3]);
      *(uint2*)(dst + (i >> 1)) = p;
    }
  }
  for (int i = tid; i < BTILE * HG_; i += 256) lds_h[i] = f2bf(0.0f);

  // per-lane geometry (M/N mapping per ISA 7.12.2; A/B use this code's contiguous k = 16*(lane/16) + s)
  const int nc  = (lane & 15) + (w << 4);   // hidden column 0..127 owned (C/D: N=lane&15)
  const int bk  = (lane < 16) ? 0 : 16;     // B operand: lanes 16-31 hold K+16
  const int ak  = (lane < 16) ? 0 : 16;     // A operand: lanes 16-31 hold K+16
  const int am  = lane & 15;                // A operand row
  const int mhi = (lane >> 4) << 3;         // C/D rows: +8 for lanes 16-31

  // ---- preload W_hh[g] B-operands into VGPRs (live across whole t-loop) ----
  v16bf Bh[3][4];
  #pragma unroll
  for (int gi = 0; gi < 3; ++gi) {
    #pragma unroll
    for (int c = 0; c < 4; ++c) {
      const float* wp = W_hh + ((size_t)g * H3_ + gi * HG_ + nc) * HG_ + 32 * c + bk;
      v16bf bb;
      #pragma unroll
      for (int e = 0; e < 16; e += 4) {
        v4f v = *(const v4f*)(wp + e);
        bb[e] = f2bf(v[0]); bb[e+1] = f2bf(v[1]);
        bb[e+2] = f2bf(v[2]); bb[e+3] = f2bf(v[3]);
      }
      Bh[gi][c] = bb;
    }
  }

  // biases (column-constant per lane)
  const float* bi = b_ih + (size_t)g * H3_;
  const float* bh = b_hh + (size_t)g * H3_;
  const float brz_r = bi[nc]            + bh[nc];
  const float brz_z = bi[HG_ + nc]      + bh[HG_ + nc];
  const float bn_x  = bi[2 * HG_ + nc];
  const float bn_h  = bh[2 * HG_ + nc];

  float hreg[8];
  #pragma unroll
  for (int i = 0; i < 8; ++i) hreg[i] = 0.0f;

  // x staging coords: thread handles row xm, 8 contiguous cols at xc
  const int xm = tid >> 4;
  const int xc = (tid & 15) << 3;
  const float* xbase = x + ((size_t)(b0 + xm) * T_) * IN_ + (size_t)g * IG_ + xc;
  uint32_t* lxu = (uint32_t*)lds_x;

  // stage x(t=0)
  {
    v4f a  = *(const v4f*)xbase;
    v4f b2 = *(const v4f*)(xbase + 4);
    uint4 p;
    p.x = pk2(a[0], a[1]);  p.y = pk2(a[2], a[3]);
    p.z = pk2(b2[0], b2[1]); p.w = pk2(b2[2], b2[3]);
    *(uint4*)(lxu + xm * 64 + (xc >> 1)) = p;
  }
  __syncthreads();

  auto loadB = [&](int jcol, int k) -> v16bf {
    const v8bf* p = (const v8bf*)&lds_wih[(size_t)jcol * IG_ + k];
    v16bf bb;
    #pragma unroll
    for (int e = 0; e < 8; ++e) { bb[e] = p[0][e]; bb[e + 8] = p[1][e]; }
    return bb;
  };

  for (int t = 0; t < T_; ++t) {
    // issue t+1 x-tile global load early (latency hidden under WMMA)
    const float* gnext = xbase + (size_t)((t + 1 < T_) ? (t + 1) : t) * IN_;
    v4f nx0 = *(const v4f*)gnext;
    v4f nx1 = *(const v4f*)(gnext + 4);
    if (t + 2 < T_) __builtin_prefetch(gnext + IN_, 0, 0); // warm L2 for t+2

    // A operands for this step: h(t) and x(t), from LDS (two b128 reads each chunk)
    v16bf Ah[4], Ax[4];
    #pragma unroll
    for (int c = 0; c < 4; ++c) {
      const int o1 = 32 * c + ak;
      v8bf l0 = *(const v8bf*)&lds_h[am * HG_ + o1];
      v8bf h0 = *(const v8bf*)&lds_h[am * HG_ + o1 + 8];
      v8bf l1 = *(const v8bf*)&lds_x[am * IG_ + o1];
      v8bf h1 = *(const v8bf*)&lds_x[am * IG_ + o1 + 8];
      #pragma unroll
      for (int e = 0; e < 8; ++e) {
        Ah[c][e] = l0[e]; Ah[c][e + 8] = h0[e];
        Ax[c][e] = l1[e]; Ax[c][e + 8] = h1[e];
      }
    }
    __syncthreads(); // all reads of lds_h / lds_x done; safe to overwrite below

    // 24 WMMAs/step: r,z fuse x-proj + h-proj in one accumulator; n kept split
    v8f cr = {}, cz = {}, cnx = {}, cnh = {};
    #pragma unroll
    for (int c = 0; c < 4; ++c) {
      const int k0 = 32 * c + bk;
      cr  = __builtin_amdgcn_wmma_f32_16x16x32_bf16(false, Ax[c], false, loadB(nc, k0),           (short)0, cr,  false, false);
      cr  = __builtin_amdgcn_wmma_f32_16x16x32_bf16(false, Ah[c], false, Bh[0][c],                (short)0, cr,  false, false);
      cz  = __builtin_amdgcn_wmma_f32_16x16x32_bf16(false, Ax[c], false, loadB(HG_ + nc, k0),     (short)0, cz,  false, false);
      cz  = __builtin_amdgcn_wmma_f32_16x16x32_bf16(false, Ah[c], false, Bh[1][c],                (short)0, cz,  false, false);
      cnx = __builtin_amdgcn_wmma_f32_16x16x32_bf16(false, Ax[c], false, loadB(2 * HG_ + nc, k0), (short)0, cnx, false, false);
      cnh = __builtin_amdgcn_wmma_f32_16x16x32_bf16(false, Ah[c], false, Bh[2][c],                (short)0, cnh, false, false);
    }

    // gates + state update + streamed output (fp32 h kept in regs)
    #pragma unroll
    for (int i = 0; i < 8; ++i) {
      const float rr = sigm(cr[i] + brz_r);
      const float zz = sigm(cz[i] + brz_z);
      const float nn = tanh_fast(cnx[i] + bn_x + rr * (cnh[i] + bn_h));
      const float hv = (1.0f - zz) * nn + zz * hreg[i];
      hreg[i] = hv;
      const int m = i + mhi;
      lds_h[m * HG_ + nc] = f2bf(hv);
      out[((size_t)(b0 + m) * T_ + t) * HID_ + (size_t)g * HG_ + nc] = hv;
    }

    // commit staged x(t+1) to LDS
    {
      uint4 p;
      p.x = pk2(nx0[0], nx0[1]); p.y = pk2(nx0[2], nx0[3]);
      p.z = pk2(nx1[0], nx1[1]); p.w = pk2(nx1[2], nx1[3]);
      *(uint4*)(lxu + xm * 64 + (xc >> 1)) = p;
    }
    __syncthreads(); // h(t+1) and x(t+1) visible for next step
  }
}

extern "C" void kernel_launch(void* const* d_in, const int* in_sizes, int n_in,
                              void* d_out, int out_size, void* d_ws, size_t ws_size,
                              hipStream_t stream) {
  (void)in_sizes; (void)n_in; (void)out_size; (void)d_ws; (void)ws_size;
  const float* x    = (const float*)d_in[0];
  const float* W_ih = (const float*)d_in[1];
  const float* W_hh = (const float*)d_in[2];
  const float* b_ih = (const float*)d_in[3];
  const float* b_hh = (const float*)d_in[4];
  float* out = (float*)d_out;

  dim3 grid((B_ / BTILE) * G_); // 32 workgroups: 4 batch tiles x 8 groups
  dim3 block(256);              // 8 wave32 waves
  grouped_gru_fused<<<grid, block, 0, stream>>>(x, W_ih, W_hh, b_ih, b_hh, out);
}
